// YakAttention_11132555231285
// MI455X (gfx1250) — matrix operand order repaired and checked by reading, not compiled, not run on hardware
//
#include <hip/hip_runtime.h>
#include <hip/hip_bf16.h>
#include <stdint.h>

typedef uint16_t u16;
typedef __attribute__((ext_vector_type(16))) __bf16 v16bf;
typedef __attribute__((ext_vector_type(8)))  __bf16 v8bf;
typedef __attribute__((ext_vector_type(8)))  float  v8f;

#define HIDDEN 4096
#define S_LEN  2048
#define NHEADS 32
#define NKV    8
#define HDIM   128

// ---------- bf16 helpers (bit-level, RNE) ----------
__device__ __forceinline__ u16 f2bf(float f) {
  union { float f; uint32_t u; } a; a.f = f;
  uint32_t u = a.u;
  uint32_t r = u + 0x7FFFu + ((u >> 16) & 1u);
  return (u16)(r >> 16);
}

__device__ __forceinline__ v16bf cat8(v8bf lo, v8bf hi) {
  return __builtin_shufflevector(lo, hi, 0,1,2,3,4,5,6,7,8,9,10,11,12,13,14,15);
}

__device__ __forceinline__ v8f wmma_bf16(v16bf a, v16bf b, v8f c) {
  // (neg_a, A, neg_b, B, c_mod, C, reuse_a, reuse_b)
  return __builtin_amdgcn_wmma_f32_16x16x32_bf16(false, a, false, b, (short)0, c,
                                                 false, false);
}

// ---------- CDNA5 async global->LDS copy (ASYNCcnt-tracked) ----------
__device__ __forceinline__ void async_ld16(uint32_t lds_off, const u16* gptr) {
  asm volatile("global_load_async_to_lds_b128 %0, %1, off"
               :: "v"(lds_off), "v"((uint64_t)(uintptr_t)gptr)
               : "memory");
}
__device__ __forceinline__ void wait_async0() {
  asm volatile("s_wait_asynccnt 0" ::: "memory");
}

// ---------- register-staged tile fragments (for GEMM double staging) ----------
struct RegF { float4 v[4]; };  // 16 f32
struct RegH { uint4  v[2]; };  // 16 bf16

__device__ __forceinline__ void greg(RegF& r, const float* p) {
#pragma unroll
  for (int i = 0; i < 4; ++i) r.v[i] = ((const float4*)p)[i];
}
__device__ __forceinline__ void greg(RegH& r, const u16* p) {
  r.v[0] = ((const uint4*)p)[0];
  r.v[1] = ((const uint4*)p)[1];
}
__device__ __forceinline__ void sreg(u16* dst, const RegF& r) {
#pragma unroll
  for (int i = 0; i < 4; ++i) {
    uint2 p;
    p.x = ((uint32_t)f2bf(r.v[i].y) << 16) | f2bf(r.v[i].x);
    p.y = ((uint32_t)f2bf(r.v[i].w) << 16) | f2bf(r.v[i].z);
    *(uint2*)(dst + 4 * i) = p;
  }
}
__device__ __forceinline__ void sreg(u16* dst, const RegH& r) {
  *(uint4*)(dst)     = r.v[0];
  *(uint4*)(dst + 8) = r.v[1];
}

template <typename T> struct RegT;
template <> struct RegT<float> { using type = RegF; };
template <> struct RegT<u16>   { using type = RegH; };

// ---------- GEMM: C[MxN] = A[MxK] * B[NxK]^T (torch-Linear layout) ----------
// Block tile 128x128, BK=32; 8 waves x (4x2) 16x16 WMMA tiles.
// Next K-chunk is register-staged during compute to hide global latency.
template <typename TA, typename TB>
__global__ __launch_bounds__(256) void gemm_nt(const TA* __restrict__ A,
                                               const TB* __restrict__ B,
                                               float* __restrict__ C,
                                               int M, int N, int K) {
  __shared__ __align__(32) u16 As[128][32];
  __shared__ __align__(32) u16 Bs[128][32];
  const int tid  = threadIdx.x;
  const int lane = tid & 31, wave = tid >> 5;
  const int wm   = wave & 1, wn = wave >> 1;   // 2 (M) x 4 (N) wave grid
  const int m0   = blockIdx.y * 128, n0 = blockIdx.x * 128;
  const int lrow = tid >> 1, lcol = (tid & 1) * 16;
  const int half = lane >> 4, mn = lane & 15;

  const TA* aptr = A + (size_t)(m0 + lrow) * K + lcol;
  const TB* bptr = B + (size_t)(n0 + lrow) * K + lcol;

  v8f acc[4][2];
#pragma unroll
  for (int i = 0; i < 4; ++i)
#pragma unroll
    for (int j = 0; j < 2; ++j)
#pragma unroll
      for (int e = 0; e < 8; ++e) acc[i][j][e] = 0.f;

  typename RegT<TA>::type ra;
  typename RegT<TB>::type rb;
  greg(ra, aptr);
  greg(rb, bptr);

  for (int kb = 0; kb < K; kb += 32) {
    sreg(&As[lrow][lcol], ra);
    sreg(&Bs[lrow][lcol], rb);
    __syncthreads();
    if (kb + 32 < K) {           // stage next chunk while this one computes
      greg(ra, aptr + kb + 32);
      greg(rb, bptr + kb + 32);
    }

    v16bf bfrag[2], afrag[4];
#pragma unroll
    for (int nt = 0; nt < 2; ++nt) { // B frag: lane = N, K = {h*8..+7, h*8+16..+23}
      const u16* bp = &Bs[wn * 32 + nt * 16 + mn][half * 8];
      bfrag[nt] = cat8(*(const v8bf*)bp, *(const v8bf*)(bp + 16));
    }
#pragma unroll
    for (int mt = 0; mt < 4; ++mt) { // A frag: lane = M, K = {h*8..+7, h*8+16..+23}
      const u16* ap = &As[wm * 64 + mt * 16 + mn][half * 8];
      afrag[mt] = cat8(*(const v8bf*)ap, *(const v8bf*)(ap + 16));
    }
#pragma unroll
    for (int mt = 0; mt < 4; ++mt)
#pragma unroll
      for (int nt = 0; nt < 2; ++nt)
        acc[mt][nt] = wmma_bf16(afrag[mt], bfrag[nt], acc[mt][nt]);
    __syncthreads();
  }

#pragma unroll
  for (int mt = 0; mt < 4; ++mt)
#pragma unroll
    for (int nt = 0; nt < 2; ++nt) {
      const int col = n0 + wn * 32 + nt * 16 + mn;
#pragma unroll
      for (int r = 0; r < 8; ++r) {
        const int row = m0 + wm * 64 + mt * 16 + r + 8 * half;
        C[(size_t)row * N + col] = acc[mt][nt][r];
      }
    }
}

// ---------- RoPE + repack to [head][s][d] bf16 ----------
__global__ void rope_pack(const float* __restrict__ P, const int* __restrict__ pos,
                          u16* __restrict__ dst, int nheads) {
  const int idx = blockIdx.x * blockDim.x + threadIdx.x;
  const int total = S_LEN * nheads * 64;
  if (idx >= total) return;
  const int i = idx & 63;
  const int h = (idx >> 6) % nheads;
  const int s = idx / (nheads * 64);
  const float* p = P + (size_t)s * nheads * HDIM + h * HDIM;
  const float x1 = p[i], x2 = p[i + 64];
  // inv_freq = 10000^(-i/64) = exp(-i * ln(10000)/64)
  const float ang = (float)pos[s] * __expf(-(float)i * (9.210340371976184f / 64.f));
  const float c = __cosf(ang), sn = __sinf(ang);
  u16* o = dst + ((size_t)h * S_LEN + s) * HDIM;
  o[i]      = f2bf(x1 * c - x2 * sn);
  o[i + 64] = f2bf(x2 * c + x1 * sn);
}

// ---------- V transpose: [s][kv*128+d] f32 -> [kv][d][s] bf16 ----------
__global__ void v_transpose(const float* __restrict__ V, u16* __restrict__ Vt) {
  const int idx = blockIdx.x * blockDim.x + threadIdx.x; // linear over Vt
  if (idx >= NKV * HDIM * S_LEN) return;
  const int s = idx & (S_LEN - 1);
  const int d = (idx >> 11) & (HDIM - 1);
  const int h = idx >> 18;
  Vt[idx] = f2bf(V[(size_t)s * (NKV * HDIM) + h * HDIM + d]);
}

// ---------- Flash attention (GQA, causal), WMMA QK^T and PV ----------
// grid (S/128, NHEADS); 8 waves each own 16 q-rows; key tiles of 32.
// K/V tiles double-buffered in LDS via CDNA5 async global->LDS copies.
__global__ __launch_bounds__(256) void flash_attn(const u16* __restrict__ Qb,
                                                  const u16* __restrict__ Kb,
                                                  const u16* __restrict__ Vt,
                                                  u16* __restrict__ Ob) {
  __shared__ __align__(32) u16 Ks[2][32][128];  // [buf][l][d]
  __shared__ __align__(32) u16 Vs[2][128][32];  // [buf][d][l]
  __shared__ __align__(32) u16 Ps[8][16][32];   // per-wave P staging
  const float scaling = 0.08838834764831845f;   // 128^-0.5

  const int tid = threadIdx.x, lane = tid & 31, wave = tid >> 5;
  const int half = lane >> 4, mn = lane & 15;
  const int head = blockIdx.y, kv = head >> 2;  // GROUPS = 4
  const int qb0 = blockIdx.x * 128;
  const int qrow = qb0 + wave * 16;

  const int kr = tid >> 3, kc = (tid & 7) * 16;  // K-tile slice per thread
  const int vr = tid >> 1, vc = (tid & 1) * 16;  // V-tile slice per thread
  const u16* kbase = Kb + (size_t)kv * S_LEN * HDIM;
  const u16* vbase = Vt + (size_t)kv * HDIM * S_LEN;

  // Q fragments (A layout), resident for whole kernel: 4 K-chunks of 32 over D=128
  v16bf qf[4];
  {
    const u16* qp = Qb + ((size_t)head * S_LEN + qrow + mn) * HDIM;
#pragma unroll
    for (int c = 0; c < 4; ++c) {
      const u16* p = qp + c * 32 + half * 8;
      qf[c] = cat8(*(const v8bf*)p, *(const v8bf*)(p + 16));
    }
  }

  float mrow[8], lrow[8];
  v8f acc[8];
#pragma unroll
  for (int r = 0; r < 8; ++r) { mrow[r] = -1e30f; lrow[r] = 0.f; }
#pragma unroll
  for (int dt = 0; dt < 8; ++dt)
#pragma unroll
    for (int e = 0; e < 8; ++e) acc[dt][e] = 0.f;

  // issue async loads for the first key tile
  async_ld16((uint32_t)(uintptr_t)&Ks[0][kr][kc],
             kbase + (size_t)(0 + kr) * HDIM + kc);
  async_ld16((uint32_t)(uintptr_t)&Vs[0][vr][vc],
             vbase + (size_t)vr * S_LEN + 0 + vc);

  const int lend = qb0 + 128;  // causal bound for this block
  for (int lb = 0; lb < lend; lb += 32) {
    const int buf = (lb >> 5) & 1;
    wait_async0();       // my async loads for `buf` are done
    __syncthreads();     // everyone's are done; everyone done reading buf^1
    if (lb + 32 < lend) {  // prefetch next tile into the other buffer
      async_ld16((uint32_t)(uintptr_t)&Ks[buf ^ 1][kr][kc],
                 kbase + (size_t)(lb + 32 + kr) * HDIM + kc);
      async_ld16((uint32_t)(uintptr_t)&Vs[buf ^ 1][vr][vc],
                 vbase + (size_t)vr * S_LEN + lb + 32 + vc);
    }

    // scores: 16 q-rows x 32 keys = two 16x16 C tiles, 4 WMMAs each over D
    v8f sc[2];
#pragma unroll
    for (int nc = 0; nc < 2; ++nc) {
#pragma unroll
      for (int e = 0; e < 8; ++e) sc[nc][e] = 0.f;
#pragma unroll
      for (int dc = 0; dc < 4; ++dc) {
        const u16* kp = &Ks[buf][nc * 16 + mn][dc * 32 + half * 8];
        v16bf kf = cat8(*(const v8bf*)kp, *(const v8bf*)(kp + 16));
        sc[nc] = wmma_bf16(qf[dc], kf, sc[nc]);
      }
    }
    // scale + causal mask (C layout: col = mn, row = r + 8*half)
#pragma unroll
    for (int nc = 0; nc < 2; ++nc) {
      const int l = lb + nc * 16 + mn;
#pragma unroll
      for (int r = 0; r < 8; ++r) {
        const int q = qrow + r + 8 * half;
        const float v = sc[nc][r] * scaling;
        sc[nc][r] = (l <= q) ? v : -1e30f;
      }
    }
    // online softmax per row (reduce across 16-lane half)
#pragma unroll
    for (int r = 0; r < 8; ++r) {
      float mx = fmaxf(sc[0][r], sc[1][r]);
#pragma unroll
      for (int off = 1; off < 16; off <<= 1)
        mx = fmaxf(mx, __shfl_xor(mx, off, 32));
      const float mnew = fmaxf(mrow[r], mx);
      const float corr = __expf(mrow[r] - mnew);
      mrow[r] = mnew;
      const float p0 = __expf(sc[0][r] - mnew);
      const float p1 = __expf(sc[1][r] - mnew);
      sc[0][r] = p0; sc[1][r] = p1;
      float rs = p0 + p1;
#pragma unroll
      for (int off = 1; off < 16; off <<= 1) rs += __shfl_xor(rs, off, 32);
      lrow[r] = lrow[r] * corr + rs;
#pragma unroll
      for (int dt = 0; dt < 8; ++dt) acc[dt][r] *= corr;
    }
    // C-layout -> A-layout for P via wave-private LDS (in-order DS per wave)
#pragma unroll
    for (int nc = 0; nc < 2; ++nc)
#pragma unroll
      for (int r = 0; r < 8; ++r)
        Ps[wave][r + 8 * half][nc * 16 + mn] = f2bf(sc[nc][r]);
    v16bf pf;
    {
      const u16* pp = &Ps[wave][mn][half * 8];
      pf = cat8(*(const v8bf*)pp, *(const v8bf*)(pp + 16));
    }
    // PV: 8 d-tiles of 16, K = 32 key positions
#pragma unroll
    for (int dt = 0; dt < 8; ++dt) {
      const u16* vp = &Vs[buf][dt * 16 + mn][half * 8];
      v16bf vf = cat8(*(const v8bf*)vp, *(const v8bf*)(vp + 16));
      acc[dt] = wmma_bf16(pf, vf, acc[dt]);
    }
  }

  // normalize + store bf16 attention output [s][head*128+d]
#pragma unroll
  for (int dt = 0; dt < 8; ++dt)
#pragma unroll
    for (int r = 0; r < 8; ++r) {
      const int row = qrow + r + 8 * half;
      const float o = acc[dt][r] / lrow[r];
      Ob[(size_t)row * (NHEADS * HDIM) + head * HDIM + dt * 16 + mn] = f2bf(o);
    }
}

// ---------- launcher ----------
extern "C" void kernel_launch(void* const* d_in, const int* in_sizes, int n_in,
                              void* d_out, int out_size, void* d_ws, size_t ws_size,
                              hipStream_t stream) {
  (void)in_sizes; (void)n_in; (void)out_size; (void)ws_size;
  const int*   pos = (const int*)  d_in[0];
  const float* X   = (const float*)d_in[1];
  const float* Wq  = (const float*)d_in[2];
  const float* Wk  = (const float*)d_in[3];
  const float* Wv  = (const float*)d_in[4];
  const float* Wo  = (const float*)d_in[5];
  float* out = (float*)d_out;

  char* ws = (char*)d_ws;
  size_t off = 0;
  auto alloc = [&](size_t bytes) -> char* {
    char* p = ws + off;
    off += (bytes + 255) & ~(size_t)255;
    return p;
  };
  float* Qf = (float*)alloc((size_t)S_LEN * HIDDEN * 4);
  float* Kf = (float*)alloc((size_t)S_LEN * NKV * HDIM * 4);
  float* Vf = (float*)alloc((size_t)S_LEN * NKV * HDIM * 4);
  u16*   Qb = (u16*)  alloc((size_t)NHEADS * S_LEN * HDIM * 2);
  u16*   Kb = (u16*)  alloc((size_t)NKV * S_LEN * HDIM * 2);
  u16*   Vt = (u16*)  alloc((size_t)NKV * HDIM * S_LEN * 2);
  u16*   Ob = (u16*)  alloc((size_t)S_LEN * HIDDEN * 2);

  dim3 blk(256);
  // projections: x @ W^T
  gemm_nt<float, float><<<dim3(HIDDEN / 128, S_LEN / 128), blk, 0, stream>>>(
      X, Wq, Qf, S_LEN, HIDDEN, HIDDEN);
  gemm_nt<float, float><<<dim3((NKV * HDIM) / 128, S_LEN / 128), blk, 0, stream>>>(
      X, Wk, Kf, S_LEN, NKV * HDIM, HIDDEN);
  gemm_nt<float, float><<<dim3((NKV * HDIM) / 128, S_LEN / 128), blk, 0, stream>>>(
      X, Wv, Vf, S_LEN, NKV * HDIM, HIDDEN);
  // RoPE + repack
  rope_pack<<<dim3((S_LEN * NHEADS * 64) / 256), blk, 0, stream>>>(Qf, pos, Qb, NHEADS);
  rope_pack<<<dim3((S_LEN * NKV * 64) / 256), blk, 0, stream>>>(Kf, pos, Kb, NKV);
  v_transpose<<<dim3((NKV * HDIM * S_LEN) / 256), blk, 0, stream>>>(Vf, Vt);
  // attention
  flash_attn<<<dim3(S_LEN / 128, NHEADS), blk, 0, stream>>>(Qb, Kb, Vt, Ob);
  // output projection -> f32 result
  gemm_nt<u16, float><<<dim3(HIDDEN / 128, S_LEN / 128), blk, 0, stream>>>(
      Ob, Wo, out, S_LEN, HIDDEN, HIDDEN);
}
